// RecurrentSimpleBody_57861799412359
// MI455X (gfx1250) — matrix operand order repaired and checked by reading, not compiled, not run on hardware
//
#include <hip/hip_runtime.h>
#include <hip/hip_bf16.h>
#include <math.h>

// ---------------------------------------------------------------------------
// Problem constants (from reference): T=1024, B=512, D=128, H=128
// ---------------------------------------------------------------------------
#define TSTEPS 1024
#define NBATCH 512
#define DDIM   128
#define HDIM   128
#define GDIM   384           // 3*H gate width

typedef __attribute__((ext_vector_type(16))) __bf16 v16bf;
typedef __attribute__((ext_vector_type(8)))  __bf16 v8bf;
typedef __attribute__((ext_vector_type(8)))  float  v8f;

__device__ __forceinline__ unsigned f2bfbits(float f) {
  unsigned u = __builtin_bit_cast(unsigned, f);
  u += 0x7FFFu + ((u >> 16) & 1u);           // round-to-nearest-even
  return u >> 16;
}

// Fast gate nonlinearities: raw v_rcp_f32 / v_exp_f32 (and native v_tanh_f32
// when the toolchain exposes it) instead of IEEE div / libm tanh.
__device__ __forceinline__ float fast_sigmoid(float x) {
  return __builtin_amdgcn_rcpf(1.f + __expf(-x));
}
__device__ __forceinline__ float fast_tanh(float x) {
#if __has_builtin(__builtin_amdgcn_tanhf)
  return __builtin_amdgcn_tanhf(x);
#else
  return 1.f - 2.f * __builtin_amdgcn_rcpf(__expf(2.f * x) + 1.f);
#endif
}

// Concatenate two contiguous 8-element bf16 runs into one A/B fragment.
__device__ __forceinline__ v16bf cat16(v8bf lo, v8bf hi) {
  return __builtin_shufflevector(lo, hi, 0,1,2,3,4,5,6,7,8,9,10,11,12,13,14,15);
}

// ---------------------------------------------------------------------------
// Kernel 0: stream-convert x (f32, [T,B,D]) -> bf16 into workspace.
// Fully parallel, bandwidth-bound: 256 MB read + 128 MB write (~16us @23.3TB/s)
// ---------------------------------------------------------------------------
__global__ void __launch_bounds__(256)
cvt_x_kernel(const float* __restrict__ x, unsigned short* __restrict__ xb) {
  const size_t i8 = ((size_t)blockIdx.x * 256 + threadIdx.x) * 8;
  const float4 f0 = *(const float4*)(x + i8);
  const float4 f1 = *(const float4*)(x + i8 + 4);
  uint4 o;
  o.x = f2bfbits(f0.x) | (f2bfbits(f0.y) << 16);
  o.y = f2bfbits(f0.z) | (f2bfbits(f0.w) << 16);
  o.z = f2bfbits(f1.x) | (f2bfbits(f1.y) << 16);
  o.w = f2bfbits(f1.z) | (f2bfbits(f1.w) << 16);
  *(uint4*)(xb + i8) = o;
}

// ---------------------------------------------------------------------------
// Kernel 1: fold fc1 into the GRU input weights.
//   Wc[g,d]  = sum_h W_ih[g,h] * W1[h,d]          (bf16 out)
//   bc[g]    = b_ih[g] + sum_h W_ih[g,h] * b1[h]  (f32 out)
//   WhhB     = bf16(W_hh)
// ---------------------------------------------------------------------------
__global__ void gru_prep_kernel(const float* __restrict__ W1,
                                const float* __restrict__ b1,
                                const float* __restrict__ W_ih,
                                const float* __restrict__ b_ih,
                                const float* __restrict__ W_hh,
                                unsigned short* __restrict__ Wc_bf,
                                unsigned short* __restrict__ Whh_bf,
                                float* __restrict__ bc) {
  const int g = blockIdx.x;
  const int d = threadIdx.x;
  float acc = 0.f;
  #pragma unroll 8
  for (int h = 0; h < 128; ++h)
    acc += W_ih[g * 128 + h] * W1[h * DDIM + d];
  Wc_bf[g * DDIM + d]  = (unsigned short)f2bfbits(acc);
  Whh_bf[g * HDIM + d] = (unsigned short)f2bfbits(W_hh[g * HDIM + d]);
  if (d == 0) {
    float s = b_ih[g];
    for (int h = 0; h < 128; ++h) s += W_ih[g * 128 + h] * b1[h];
    bc[g] = s;
  }
}

// ---------------------------------------------------------------------------
// Kernel 2: persistent batch-parallel GRU scan.
// grid = 32 blocks x 16 batch rows; block = 256 threads = 8 waves.
// Wave w owns gate column-tiles {w, w+8, w+16} (the r/z/n slices of hidden
// columns [16w,16w+16)). r/z slices are computed as one logical K=256 GEMM
// (GEMM2 accumulates onto GEMM1's C registers); only the n slice is split.
// h lives in f32 registers; a 4KB bf16 LDS shadow is GEMM2's A matrix.
// All 24 B fragments (Wc, W_hh tiles) are loop-invariant registers.
// x tiles for step t+2 are prefetched into cache to hide HBM latency across
// the serial recurrence.
// ---------------------------------------------------------------------------
__global__ void __launch_bounds__(256)
gru_scan_kernel(const unsigned short* __restrict__ xb_u16,
                const unsigned short* __restrict__ Wc_u16,
                const unsigned short* __restrict__ Whh_u16,
                const float* __restrict__ bc,
                const float* __restrict__ b_hh,
                float* __restrict__ out) {
  __shared__ unsigned short s_hb[16][HDIM];   // bf16 shadow of h (A-matrix src)

  const int tid     = threadIdx.x;
  const int lane    = tid & 31;
  const int wave    = tid >> 5;
  const int rowBase = blockIdx.x * 16;

  const int m15    = lane & 15;
  const int half8  = (lane >> 4) << 3;   // 0 or 8   (A lane K-offset, C row off)
  const int half16 = (lane >> 4) << 4;   // 0 or 16  (unused; B K-offset is half8)

  const __bf16* xb  = (const __bf16*)xb_u16;
  const __bf16* Wc  = (const __bf16*)Wc_u16;
  const __bf16* Whh = (const __bf16*)Whh_u16;

  // ---- loop-invariant B fragments: B[k,n] = W[n,k] -> contiguous in k ----
  // g=0,1,2 selects the r/z/n gate slice; column tile row = wave*16+g*128+m15.
  v16bf Bc[3][4], Bh[3][4];
  #pragma unroll
  for (int g = 0; g < 3; ++g) {
    const int nrow = wave * 16 + g * 128 + m15;
    #pragma unroll
    for (int kc = 0; kc < 4; ++kc) {
      const int ko = kc * 32 + half8;
      Bc[g][kc] = cat16(*(const v8bf*)(Wc  + (size_t)nrow * DDIM + ko), *(const v8bf*)(Wc  + (size_t)nrow * DDIM + ko + 16));
      Bh[g][kc] = cat16(*(const v8bf*)(Whh + (size_t)nrow * HDIM + ko), *(const v8bf*)(Whh + (size_t)nrow * HDIM + ko + 16));
    }
  }
  // ---- per-lane bias scalars (column N = wave*16 + m15 fixed per lane) ----
  const int   ncol = wave * 16 + m15;
  const float br   = bc[ncol]       + b_hh[ncol];          // r: folded
  const float bz   = bc[ncol + 128] + b_hh[ncol + 128];    // z: folded
  const float bcn  = bc[ncol + 256];                       // n: input side
  const float bhn  = b_hh[ncol + 256];                     // n: hidden side

  // ---- h = 0 (registers + LDS shadow) ----
  v8f h = {};
  for (int i = tid; i < 16 * HDIM; i += 256) (&s_hb[0][0])[i] = 0;
  __syncthreads();

  const __bf16* xrow0 = xb + (size_t)(rowBase + m15) * DDIM;

  for (int t = 0; t < TSTEPS; ++t) {
    // -------- prefetch x tile for step t+2 (hides HBM latency) -------------
    {
      const __bf16* xpre = xrow0 + (size_t)(t + 2) * NBATCH * DDIM;
      __builtin_prefetch(xpre, 0, 3);
      __builtin_prefetch(xpre + 64, 0, 3);   // second half of the 256B row
    }

    // -------- GEMM 1: gates_i = x_t @ Wc^T  (A = pre-converted bf16 x) -----
    v16bf ax[4];
    {
      const __bf16* xrow = xrow0 + (size_t)t * NBATCH * DDIM;
      #pragma unroll
      for (int kc = 0; kc < 4; ++kc) {
        const int ko = kc * 32 + half8;
        ax[kc] = cat16(*(const v8bf*)(xrow + ko), *(const v8bf*)(xrow + ko + 16));
      }
    }
    v8f accr = {}, accz = {}, accgn = {}, acchn = {};
    #pragma unroll
    for (int kc = 0; kc < 4; ++kc) {
      accr  = __builtin_amdgcn_wmma_f32_16x16x32_bf16(
                  false, ax[kc], false, Bc[0][kc], (short)0, accr,  false, false);
      accz  = __builtin_amdgcn_wmma_f32_16x16x32_bf16(
                  false, ax[kc], false, Bc[1][kc], (short)0, accz,  false, false);
      accgn = __builtin_amdgcn_wmma_f32_16x16x32_bf16(
                  false, ax[kc], false, Bc[2][kc], (short)0, accgn, false, false);
    }

    // -------- GEMM 2: += h @ W_hh^T (r/z accumulate in place; n split) -----
    v16bf ah[4];
    {
      const __bf16* hrow = (const __bf16*)&s_hb[m15][0];
      #pragma unroll
      for (int kc = 0; kc < 4; ++kc) {
        const int ko = kc * 32 + half8;
        ah[kc] = cat16(*(const v8bf*)(hrow + ko), *(const v8bf*)(hrow + ko + 16));
      }
    }
    __syncthreads();   // all s_hb reads complete before this step's writes
    #pragma unroll
    for (int kc = 0; kc < 4; ++kc) {
      accr  = __builtin_amdgcn_wmma_f32_16x16x32_bf16(
                  false, ah[kc], false, Bh[0][kc], (short)0, accr,  false, false);
      accz  = __builtin_amdgcn_wmma_f32_16x16x32_bf16(
                  false, ah[kc], false, Bh[1][kc], (short)0, accz,  false, false);
      acchn = __builtin_amdgcn_wmma_f32_16x16x32_bf16(
                  false, ah[kc], false, Bh[2][kc], (short)0, acchn, false, false);
    }

    // -------- fused gate math in C-register layout; update h ---------------
    #pragma unroll
    for (int j = 0; j < 8; ++j) {
      const float rg = fast_sigmoid(accr[j] + br);
      const float zg = fast_sigmoid(accz[j] + bz);
      const float ng = fast_tanh(accgn[j] + bcn + rg * (acchn[j] + bhn));
      h[j] = (1.f - zg) * ng + zg * h[j];
      s_hb[j + half8][ncol] = (unsigned short)f2bfbits(h[j]);
    }
    __syncthreads();   // writes visible before next iteration's A_h reads
  }

  // ---- final hidden state -> d_out [B, H] f32 (C-layout scatter) ----------
  #pragma unroll
  for (int j = 0; j < 8; ++j)
    out[(size_t)(rowBase + j + half8) * HDIM + ncol] = h[j];
}

// ---------------------------------------------------------------------------
// Launch wrapper. Inputs (setup_inputs order):
//   0:x [T,B,D] f32  1:W1 [128,D] f32  2:b1 [128] f32
//   3:W_ih [384,128] f32  4:W_hh [384,128] f32  5:b_ih [384] f32 6:b_hh [384] f32
// Output: h_final [B,H] f32.
// Workspace: [0,128M) x bf16 | [+0,96K) Wc bf16 | [+96K,192K) Whh bf16 | bc f32
// ---------------------------------------------------------------------------
extern "C" void kernel_launch(void* const* d_in, const int* in_sizes, int n_in,
                              void* d_out, int out_size, void* d_ws, size_t ws_size,
                              hipStream_t stream) {
  (void)in_sizes; (void)n_in; (void)out_size; (void)ws_size;
  const float* x    = (const float*)d_in[0];
  const float* W1   = (const float*)d_in[1];
  const float* b1   = (const float*)d_in[2];
  const float* W_ih = (const float*)d_in[3];
  const float* W_hh = (const float*)d_in[4];
  const float* b_ih = (const float*)d_in[5];
  const float* b_hh = (const float*)d_in[6];

  const size_t xbBytes = (size_t)TSTEPS * NBATCH * DDIM * 2;   // 128 MB
  char* ws = (char*)d_ws;
  unsigned short* xb     = (unsigned short*)(ws);
  unsigned short* Wc_bf  = (unsigned short*)(ws + xbBytes);
  unsigned short* Whh_bf = (unsigned short*)(ws + xbBytes + 98304);
  float*          bc     = (float*)(ws + xbBytes + 196608);

  const size_t nElem = (size_t)TSTEPS * NBATCH * DDIM;         // 64M
  cvt_x_kernel<<<dim3((unsigned)(nElem / (256 * 8))), dim3(256), 0, stream>>>(x, xb);

  gru_prep_kernel<<<dim3(GDIM), dim3(128), 0, stream>>>(
      W1, b1, W_ih, b_ih, W_hh, Wc_bf, Whh_bf, bc);

  gru_scan_kernel<<<dim3(NBATCH / 16), dim3(256), 0, stream>>>(
      xb, Wc_bf, Whh_bf, bc, b_hh, (float*)d_out);
}
